// H2GCNConv_936302871073
// MI455X (gfx1250) — matrix operand order repaired and checked by reading, not compiled, not run on hardware
//
#include <hip/hip_runtime.h>

typedef _Float16 h8   __attribute__((ext_vector_type(8)));
typedef _Float16 v16h __attribute__((ext_vector_type(16)));
typedef float    v8f  __attribute__((ext_vector_type(8)));

#define D_IN  128
#define D_CAT 384
#define D_OUT 128

// ---------------- utility kernels ----------------

__global__ void h2gcn_zero_f32(float* __restrict__ p, int n) {
    int t = blockIdx.x * blockDim.x + threadIdx.x;
    if (t < n) p[t] = 0.0f;
}

__global__ void h2gcn_degree(const int* __restrict__ ei, int E,
                             float* __restrict__ deg) {
    int e = blockIdx.x * blockDim.x + threadIdx.x;
    if (e >= E) return;
    int d = ei[E + e];   // edge_index[1] = dst
    __hip_atomic_fetch_add(&deg[d], 1.0f, __ATOMIC_RELAXED, __HIP_MEMORY_SCOPE_AGENT);
}

// cast x (f32 [N,128]) into feats f16 columns [0,128)
__global__ void h2gcn_cast_x(const float* __restrict__ x,
                             _Float16* __restrict__ feats, int n) {
    int t = blockIdx.x * blockDim.x + threadIdx.x;
    if (t >= n) return;
    int node = t >> 7, c = t & 127;
    feats[(size_t)node * D_CAT + c] = (_Float16)x[t];
}

__global__ void h2gcn_cast_w(const float* __restrict__ W,
                             _Float16* __restrict__ Wf16, int n) {
    int t = blockIdx.x * blockDim.x + threadIdx.x;
    if (t < n) Wf16[t] = (_Float16)W[t];
}

// one wave (32 lanes) per edge; each lane moves float4 (32 lanes * 4 = 128 cols)
__global__ void h2gcn_scatter(const float* __restrict__ feat,
                              const int* __restrict__ ei, int E,
                              float* __restrict__ sum) {
    unsigned t = blockIdx.x * blockDim.x + threadIdx.x;
    unsigned e = t >> 5;
    int lane = t & 31;
    if (e >= (unsigned)E) return;
    int s = ei[e];
    int d = ei[E + e];
    const float4 v = *(const float4*)(feat + (size_t)s * D_IN + lane * 4);
    float* p = sum + (size_t)d * D_IN + lane * 4;
    __hip_atomic_fetch_add(p + 0, v.x, __ATOMIC_RELAXED, __HIP_MEMORY_SCOPE_AGENT);
    __hip_atomic_fetch_add(p + 1, v.y, __ATOMIC_RELAXED, __HIP_MEMORY_SCOPE_AGENT);
    __hip_atomic_fetch_add(p + 2, v.z, __ATOMIC_RELAXED, __HIP_MEMORY_SCOPE_AGENT);
    __hip_atomic_fetch_add(p + 3, v.w, __ATOMIC_RELAXED, __HIP_MEMORY_SCOPE_AGENT);
}

// hop = sum / max(deg,1); optionally keep f32 copy (needed for next hop's gather),
// always write f16 into feats at column offset colOff
__global__ void h2gcn_finalize(const float* __restrict__ sum,
                               const float* __restrict__ deg,
                               float* __restrict__ hopF32,      // may be null
                               _Float16* __restrict__ feats,
                               int colOff, int n) {
    int t = blockIdx.x * blockDim.x + threadIdx.x;
    if (t >= n) return;
    int node = t >> 7, c = t & 127;
    float h = sum[t] / fmaxf(deg[node], 1.0f);
    if (hopF32) hopF32[t] = h;
    feats[(size_t)node * D_CAT + colOff + c] = (_Float16)h;
}

// ---------------- WMMA GEMM: out[N,128] = feats[N,384] @ Wf16[128,384]^T + bias ----------------
// One wave per 16-row strip; accumulates all 8 output 16x16 tiles (K loop of 12x32).
__global__ __launch_bounds__(256)
void h2gcn_gemm_wmma(const _Float16* __restrict__ feats,
                     const _Float16* __restrict__ Wf16,
                     const float* __restrict__ bias,
                     float* __restrict__ out, int N) {
    const int wave = (blockIdx.x * blockDim.x + threadIdx.x) >> 5;
    const int lane = threadIdx.x & 31;
    const int row0 = wave << 4;
    if (row0 >= N) return;                 // wave-uniform: EXEC stays all-1s

    const int hi    = lane >> 4;           // half-wave select
    const int r     = row0 + (lane & 15);  // A row for this lane
    const int oLane = lane & 15;           // output column within 16-wide tile

    const _Float16* arow = feats + (size_t)r * D_CAT;

    v8f acc[8] = {};

#pragma unroll
    for (int k0 = 0; k0 < D_CAT; k0 += 32) {
        // A 16x32 in contiguous k order (same as B): lanes 0-15 hold K 0..15, lanes 16-31 hold K 16..31
        h8 alo = *(const h8*)(arow + k0 + hi * 16);
        h8 ahi = *(const h8*)(arow + k0 + hi * 16 + 8);
        v16h a = __builtin_shufflevector(alo, ahi,
                                         0, 1, 2, 3, 4, 5, 6, 7,
                                         8, 9, 10, 11, 12, 13, 14, 15);
        // B 32x16 in contiguous k order (same as A): lanes 0-15 hold K 0..15, lanes 16-31 hold K 16..31
        // B[k][o] = Wf16[o][k] -> one contiguous 32B load per lane per tile
        const _Float16* wbase = Wf16 + (size_t)oLane * D_CAT + k0 + hi * 16;
#pragma unroll
        for (int o = 0; o < 8; ++o) {
            v16h bfrag = *(const v16h*)(wbase + (size_t)o * 16 * D_CAT);
            acc[o] = __builtin_amdgcn_wmma_f32_16x16x32_f16(
                false, a, false, bfrag, (short)0, acc[o], false, false);
        }
    }

    // 32-bit C/D 16x16 layout: VGPR v -> row (v + 8*hi), col = oLane
    const int rbase = row0 + hi * 8;
#pragma unroll
    for (int o = 0; o < 8; ++o) {
        const int c = o * 16 + oLane;
        const float bb = bias[c];
#pragma unroll
        for (int v = 0; v < 8; ++v) {
            out[(size_t)(rbase + v) * D_OUT + c] = acc[o][v] + bb;
        }
    }
}

// ---------------- host launcher ----------------

extern "C" void kernel_launch(void* const* d_in, const int* in_sizes, int n_in,
                              void* d_out, int out_size, void* d_ws, size_t ws_size,
                              hipStream_t stream) {
    const float* x    = (const float*)d_in[0];
    const int*   ei   = (const int*)d_in[1];
    const float* W    = (const float*)d_in[2];
    const float* bias = (const float*)d_in[3];
    float*       out  = (float*)d_out;

    const int N = in_sizes[0] / D_IN;
    const int E = in_sizes[1] / 2;
    const int ND = N * D_IN;          // N*128 (fits int)
    const int WN = D_OUT * D_CAT;     // 49152

    // workspace carve-out (256B aligned)
    char* ws = (char*)d_ws;
    size_t off = 0;
    auto carve = [&](size_t bytes) -> void* {
        void* p = ws + off;
        off = (off + bytes + 255) & ~(size_t)255;
        return p;
    };
    float*     deg   = (float*)carve((size_t)N * 4);
    float*     sum   = (float*)carve((size_t)ND * 4);
    float*     hop1  = (float*)carve((size_t)ND * 4);
    _Float16*  feats = (_Float16*)carve((size_t)N * D_CAT * 2);
    _Float16*  Wf16  = (_Float16*)carve((size_t)WN * 2);

    const int T = 256;
    auto cdiv = [](unsigned a, unsigned b) { return (a + b - 1) / b; };

    // init
    h2gcn_zero_f32<<<cdiv(N, T), T, 0, stream>>>(deg, N);
    h2gcn_zero_f32<<<cdiv(ND, T), T, 0, stream>>>(sum, ND);
    h2gcn_degree<<<cdiv(E, T), T, 0, stream>>>(ei, E, deg);
    h2gcn_cast_x<<<cdiv(ND, T), T, 0, stream>>>(x, feats, ND);
    h2gcn_cast_w<<<cdiv(WN, T), T, 0, stream>>>(W, Wf16, WN);

    // hop 1
    h2gcn_scatter<<<cdiv((unsigned)E * 32u, T), T, 0, stream>>>(x, ei, E, sum);
    h2gcn_finalize<<<cdiv(ND, T), T, 0, stream>>>(sum, deg, hop1, feats, D_IN, ND);

    // hop 2
    h2gcn_zero_f32<<<cdiv(ND, T), T, 0, stream>>>(sum, ND);
    h2gcn_scatter<<<cdiv((unsigned)E * 32u, T), T, 0, stream>>>(hop1, ei, E, sum);
    h2gcn_finalize<<<cdiv(ND, T), T, 0, stream>>>(sum, deg, nullptr, feats, 2 * D_IN, ND);

    // GEMM: one wave per 16 rows, 8 waves per block
    const unsigned nWaves  = cdiv(N, 16);
    const unsigned nBlocks = cdiv(nWaves, 8);
    h2gcn_gemm_wmma<<<nBlocks, 256, 0, stream>>>(feats, Wf16, bias, out, N);
}
